// SAE_49048526520980
// MI455X (gfx1250) — matrix operand order repaired and checked by reading, not compiled, not run on hardware
//
#include <hip/hip_runtime.h>

// SAE forward for MI455X (gfx1250, wave32):
//   encode GEMM in f16 WMMA (f32 accum), fused streaming top-64, sparse decode.
// N=8192, D=768, W=24576, K=64
// v2: 32x32 register tile per wave (2x2 WMMA, A&B reuse), ROWS=64 per WG,
//     global_prefetch of next B tile.

typedef _Float16 v16h __attribute__((ext_vector_type(16)));
typedef _Float16 v8h  __attribute__((ext_vector_type(8)));
typedef float    v8f  __attribute__((ext_vector_type(8)));

#define SAE_N    8192
#define SAE_D    768
#define SAE_W    24576
#define SAE_K    64
#define ROWS     64      // rows per workgroup
#define CT       128     // column tile streamed per iteration
#define LSTR     132     // padded LDS row stride (dwords): 132 % 64 = 4 -> conflict-free spills

// ---------------- wave32 shuffle helpers (ds_bpermute: byte index, bits [6:2] used) ---------
__device__ __forceinline__ float shfl_f(float v, int srcLane) {
  return __int_as_float(__builtin_amdgcn_ds_bpermute(srcLane << 2, __float_as_int(v)));
}
__device__ __forceinline__ float shflx_f(float v, int m, int lane) {
  return __int_as_float(__builtin_amdgcn_ds_bpermute((lane ^ m) << 2, __float_as_int(v)));
}
__device__ __forceinline__ int shflx_i(int v, int m, int lane) {
  return __builtin_amdgcn_ds_bpermute((lane ^ m) << 2, v);
}

// ---------------- prep kernels --------------------------------------------------------------
__global__ void sae_prep_x(const float* __restrict__ x, const float* __restrict__ bd,
                           _Float16* __restrict__ xh) {
  const int row = blockIdx.x;
  const float* xr = x + (size_t)row * SAE_D;
  _Float16* xo = xh + (size_t)row * SAE_D;
  for (int c = threadIdx.x; c < SAE_D; c += 256)
    xo[c] = (_Float16)(xr[c] - bd[c]);
}

__global__ void sae_prep_ae(const float* __restrict__ ae, _Float16* __restrict__ aeh) {
  const size_t total = (size_t)SAE_W * SAE_D;
  const size_t stride = (size_t)gridDim.x * blockDim.x;
  for (size_t i = (size_t)blockIdx.x * blockDim.x + threadIdx.x; i < total; i += stride)
    aeh[i] = (_Float16)ae[i];
}

// Ad [D, W] -> AdT [W, D]   (32x32 LDS tiles)
__global__ void sae_transpose_ad(const float* __restrict__ Ad, float* __restrict__ AdT) {
  __shared__ float t[32][33];
  const int tx = threadIdx.x, ty = threadIdx.y;         // block (32, 8)
  const int jt = blockIdx.x * 32, dt = blockIdx.y * 32;
#pragma unroll
  for (int i = 0; i < 4; ++i)
    t[ty + i * 8][tx] = Ad[(size_t)(dt + ty + i * 8) * SAE_W + jt + tx];
  __syncthreads();
#pragma unroll
  for (int i = 0; i < 4; ++i)
    AdT[(size_t)(jt + ty + i * 8) * SAE_D + dt + tx] = t[tx][ty + i * 8];
}

// ---------------- main fused kernel ---------------------------------------------------------
__global__ __launch_bounds__(256)
void sae_main(const _Float16* __restrict__ xh, const _Float16* __restrict__ aeh,
              const float* __restrict__ be, const float* __restrict__ adT,
              const float* __restrict__ bd, float* __restrict__ out) {
  __shared__ float lds_pre[ROWS][LSTR];   // streamed pre-activation tile (f32)
  __shared__ float lds_tv[ROWS][SAE_K];   // final top-k values
  __shared__ int   lds_ti[ROWS][SAE_K];   // final top-k indices

  const int tid  = threadIdx.x;
  const int lane = tid & 31;
  const int wid  = tid >> 5;              // 8 waves
  const int row0 = blockIdx.x * ROWS;

  // GEMM tiling: wave -> 32-row group (rg) x 32-col group (cg) = 2x2 WMMA tiles
  const int rg = wid & 1;                 // row group: rows rg*32 .. rg*32+31
  const int cg = wid >> 1;                // col group: cols cg*32 .. cg*32+31 (within CT)
  const int R0 = rg * 32;
  const int C0 = cg * 32;
  const int m  = lane & 15;
  const int h  = lane >> 4;

  const _Float16* aRow0 = xh + (size_t)(row0 + R0 + m) * SAE_D;       // A tile rows R0..R0+15
  const _Float16* aRow1 = aRow0 + (size_t)16 * SAE_D;                 // A tile rows R0+16..+31

  // streaming top-64 state: 64 entries/row = 2 per lane, 8 rows per wave
  float kv[8][2]; int ki[8][2]; float thr[8];
#pragma unroll
  for (int q = 0; q < 8; ++q) {
    kv[q][0] = kv[q][1] = -__builtin_inff();
    ki[q][0] = ki[q][1] = 0;
    thr[q]   = -__builtin_inff();
  }

  for (int ct = 0; ct < SAE_W / CT; ++ct) {
    const int c0 = ct * CT;
    const _Float16* bCol0 = aeh + (size_t)(c0 + C0 + m) * SAE_D;      // B tile cols C0..C0+15
    const _Float16* bCol1 = bCol0 + (size_t)16 * SAE_D;               // B tile cols C0+16..+31

    // prefetch next column tile's B rows (one wave of the rg pair; 12 lines cover 768 f16)
    if (rg == 0 && ct + 1 < SAE_W / CT) {
      const _Float16* nb = aeh + (size_t)(c0 + CT + C0 + lane) * SAE_D;
#pragma unroll
      for (int pf = 0; pf < 12; ++pf)
        __builtin_prefetch(nb + pf * 64, 0, 3);       // 64 halves = one 128B line
    }

    v8f acc00 = {}; v8f acc01 = {}; v8f acc10 = {}; v8f acc11 = {};
#pragma unroll 2
    for (int kt = 0; kt < SAE_D / 32; ++kt) {
      const int k0 = kt * 32;
      // A 16x32 f16 fragment: lane h=0 -> K {0..7, 16..23}; h=1 -> K {8..15, 24..31}
      v8h a0lo = *(const v8h*)(aRow0 + k0 + h * 8);
      v8h a0hi = *(const v8h*)(aRow0 + k0 + h * 8 + 16);
      v8h a1lo = *(const v8h*)(aRow1 + k0 + h * 8);
      v8h a1hi = *(const v8h*)(aRow1 + k0 + h * 8 + 16);
      v16h a0 = __builtin_shufflevector(a0lo, a0hi, 0,1,2,3,4,5,6,7,8,9,10,11,12,13,14,15);
      v16h a1 = __builtin_shufflevector(a1lo, a1hi, 0,1,2,3,4,5,6,7,8,9,10,11,12,13,14,15);
      // B 32x16 fragment: lane n holds K {k0+h*8..+7, k0+16+h*8..+7} from Ae row (c+n)
      v8h b0lo = *(const v8h*)(bCol0 + k0 + h * 8);
      v8h b0hi = *(const v8h*)(bCol0 + k0 + h * 8 + 16);
      v8h b1lo = *(const v8h*)(bCol1 + k0 + h * 8);
      v8h b1hi = *(const v8h*)(bCol1 + k0 + h * 8 + 16);
      v16h b0 = __builtin_shufflevector(b0lo, b0hi, 0,1,2,3,4,5,6,7,8,9,10,11,12,13,14,15);
      v16h b1 = __builtin_shufflevector(b1lo, b1hi, 0,1,2,3,4,5,6,7,8,9,10,11,12,13,14,15);
      acc00 = __builtin_amdgcn_wmma_f32_16x16x32_f16(false, a0, false, b0, (short)0, acc00, false, false);
      acc01 = __builtin_amdgcn_wmma_f32_16x16x32_f16(false, a0, false, b1, (short)0, acc01, false, false);
      acc10 = __builtin_amdgcn_wmma_f32_16x16x32_f16(false, a1, false, b0, (short)0, acc10, false, false);
      acc11 = __builtin_amdgcn_wmma_f32_16x16x32_f16(false, a1, false, b1, (short)0, acc11, false, false);
    }

    // add encoder bias, spill C tiles to LDS (padded stride -> conflict-free)
    const float be0 = be[c0 + C0 + m];
    const float be1 = be[c0 + C0 + 16 + m];
    const int rb0 = R0 + (h ? 8 : 0);
    const int rb1 = R0 + 16 + (h ? 8 : 0);
#pragma unroll
    for (int r = 0; r < 8; ++r) {
      lds_pre[rb0 + r][C0 + m]      = acc00[r] + be0;
      lds_pre[rb0 + r][C0 + 16 + m] = acc01[r] + be1;
      lds_pre[rb1 + r][C0 + m]      = acc10[r] + be0;
      lds_pre[rb1 + r][C0 + 16 + m] = acc11[r] + be1;
    }
    __syncthreads();

    // streaming exact top-64: wave owns rows wid*8 .. wid*8+7
#pragma unroll
    for (int q = 0; q < 8; ++q) {
      const int row = wid * 8 + q;
      for (int b = 0; b < CT / 32; ++b) {
        const float val = lds_pre[row][b * 32 + lane];
        unsigned mask = __builtin_amdgcn_ballot_w32(val > thr[q]);
        while (mask) {
          const int src = __builtin_ctz(mask);
          mask &= mask - 1;
          const float v = shfl_f(val, src);       // uniform broadcast
          if (v > thr[q]) {                       // thr may have risen: re-check (uniform)
            // per-lane min of 2 slots, payload = (lane<<1)|slot
            float lm; int lp;
            if (kv[q][1] < kv[q][0]) { lm = kv[q][1]; lp = (lane << 1) | 1; }
            else                     { lm = kv[q][0]; lp = (lane << 1); }
#pragma unroll
            for (int off = 16; off > 0; off >>= 1) {
              const float om = shflx_f(lm, off, lane);
              const int   op = shflx_i(lp, off, lane);
              if (om < lm || (om == lm && op < lp)) { lm = om; lp = op; }
            }
            if ((lp >> 1) == lane) {              // owner replaces its min slot
              if (lp & 1) { kv[q][1] = v; ki[q][1] = c0 + b * 32 + src; }
              else        { kv[q][0] = v; ki[q][0] = c0 + b * 32 + src; }
            }
            float nm = fminf(kv[q][0], kv[q][1]); // new threshold = new global min
#pragma unroll
            for (int off = 16; off > 0; off >>= 1)
              nm = fminf(nm, shflx_f(nm, off, lane));
            thr[q] = nm;
          }
        }
      }
    }
    __syncthreads();
  }

  // publish top-k lists
#pragma unroll
  for (int q = 0; q < 8; ++q) {
    const int row = wid * 8 + q;
    lds_tv[row][lane]      = kv[q][0];
    lds_tv[row][lane + 32] = kv[q][1];
    lds_ti[row][lane]      = ki[q][0];
    lds_ti[row][lane + 32] = ki[q][1];
  }
  __syncthreads();

  // sparse decode: out[n,:] = bd + sum relu(v_e) * AdT[idx_e,:]   (AdT resident in L2)
  const int cA = tid, cB = tid + 256, cC = tid + 512;
  const float bdA = bd[cA], bdB = bd[cB], bdC = bd[cC];
  for (int r = 0; r < ROWS; ++r) {
    float a0 = bdA, a1 = bdB, a2 = bdC;
    for (int e = 0; e < SAE_K; ++e) {
      const float v = lds_tv[r][e];               // uniform across block
      if (v > 0.0f) {
        const float* ad = adT + (size_t)lds_ti[r][e] * SAE_D;
        a0 = fmaf(v, ad[cA], a0);
        a1 = fmaf(v, ad[cB], a1);
        a2 = fmaf(v, ad[cC], a2);
      }
    }
    float* o = out + (size_t)(row0 + r) * SAE_D;
    o[cA] = a0; o[cB] = a1; o[cC] = a2;
  }
}

// ---------------- host launch ---------------------------------------------------------------
extern "C" void kernel_launch(void* const* d_in, const int* in_sizes, int n_in,
                              void* d_out, int out_size, void* d_ws, size_t ws_size,
                              hipStream_t stream) {
  const float* x  = (const float*)d_in[0];   // [N, D]
  const float* Ae = (const float*)d_in[1];   // [W, D]
  const float* be = (const float*)d_in[2];   // [W]
  const float* Ad = (const float*)d_in[3];   // [D, W]
  const float* bd = (const float*)d_in[4];   // [D]
  float* out = (float*)d_out;                // [N, D]

  char* ws = (char*)d_ws;
  _Float16* xh  = (_Float16*)(ws);                          // N*D*2  = 12,582,912 B
  _Float16* aeh = (_Float16*)(ws + 12582912);               // W*D*2  = 37,748,736 B
  float*    adT = (float*)   (ws + 12582912 + 37748736);    // W*D*4  = 75,497,472 B

  sae_prep_x<<<SAE_N, 256, 0, stream>>>(x, bd, xh);
  sae_prep_ae<<<4608, 256, 0, stream>>>(Ae, aeh);
  sae_transpose_ad<<<dim3(SAE_W / 32, SAE_D / 32), dim3(32, 8), 0, stream>>>(Ad, adT);
  sae_main<<<SAE_N / ROWS, 256, 0, stream>>>(xh, aeh, be, adT, bd, out);
}
